// Spatial_Memory_4939212390857
// MI455X (gfx1250) — compile-verified, matrix operand order checked by reading, not run on hardware
//
#include <hip/hip_runtime.h>
#include <hip/hip_bf16.h>

#define B_   512
#define C_   256
#define VD_  256
#define HW_  225
#define NT_  15     // n row tiles (15*16 = 240 >= 225)

typedef __attribute__((ext_vector_type(16))) __bf16 v16bf;
typedef __attribute__((ext_vector_type(8)))  float  v8f;

union BF16Frag {
    v16bf v;
    unsigned short u[16];
    unsigned d[8];
    uint4 q[2];
};

// round-to-nearest (ties away) f32 -> bf16, single element
__device__ __forceinline__ unsigned short f2bf_bits(float x) {
    union { float f; unsigned u; } c; c.f = x;
    return (unsigned short)((c.u + 0x8000u) >> 16);
}

// pack two f32 -> {bf16(hi),bf16(lo)} dword via v_perm_b32 (3 VALU ops total)
__device__ __forceinline__ unsigned pack_bf2(float lo, float hi) {
    union { float f; unsigned u; } a, b; a.f = lo; b.f = hi;
    unsigned x = a.u + 0x8000u;
    unsigned y = b.u + 0x8000u;
    // result bytes: [x.b2, x.b3, y.b2, y.b3]  ->  (y>>16)<<16 | (x>>16)
    return __builtin_amdgcn_perm(y, x, 0x07060302u);
}

__global__ __launch_bounds__(256) void spatial_memory_kernel(
    const float* __restrict__ m_in,   // [B,225,256]
    const float* __restrict__ m_out,  // [B,225,256]
    const float* __restrict__ q_in,   // [B,256,225]
    const float* __restrict__ q_out,  // [B,256,225]
    float* __restrict__ out)          // [B,512,225] mem_out, then [B,225,225] p
{
    const int mt   = blockIdx.x;      // 0..14  (16-wide m column tile)
    const int b    = blockIdx.y;      // 0..511
    const int t    = threadIdx.x;     // 0..255
    const int lane = t & 31;
    const int wv   = t >> 5;          // 0..7
    const int col16 = lane & 15;
    const int g     = lane >> 4;      // K-half selector in WMMA layouts
    const int m0    = mt * 16;

    // bf16 operand tiles stored [outer][k] with padded stride: frag reads = 2x ds_load_b128
    __shared__ unsigned short Bq [16 * 264];   // q_in tile   [col][k],   K=C=256
    __shared__ unsigned short Pbf[16 * 264];   // p tile      [col][n],   K padded to 256
    __shared__ unsigned short Slab[256 * 40];  // m_out k-slab [v][n_loc], 32 n + 8 pad
    __shared__ float S[240 * 16];              // pre-softmax scores [n][col]
    __shared__ float red[2][16][16];           // softmax reductions [chunk][col]
    __shared__ float dtab[32];                 // 0.92^d, d=0..28

    if (t < 29) dtab[t] = __powf(0.92f, (float)t);
    for (int i = t; i < 16 * 264; i += 256) Pbf[i] = 0;

    // ---- stage q_in[b][:, m0:m0+16] -> LDS bf16 (pair-packed dword stores) ----
    {
        const float* qb = q_in + (size_t)b * C_ * HW_;
        const int col = t & 15, kp0 = t >> 4;
        const int m = m0 + col;
        const bool mok = (m < HW_);
        #pragma unroll
        for (int kk = 0; kk < 8; ++kk) {
            const int pi = kp0 + kk * 16;          // k-pair index 0..127
            const int k = 2 * pi;
            float x0 = mok ? qb[k * HW_ + m]       : 0.0f;
            float x1 = mok ? qb[(k + 1) * HW_ + m] : 0.0f;
            *(unsigned*)&Bq[col * 264 + 2 * pi] = pack_bf2(x0, x1);
        }
    }

    // ---- q_out passthrough -> out channels 256..511 (pure streaming) ----
    {
        const float* qo  = q_out + (size_t)b * VD_ * HW_;
        float*       dst = out + (size_t)b * 512 * HW_ + (size_t)256 * HW_;
        const int col = t & 15, c0 = t >> 4;
        const int m = m0 + col;
        if (m < HW_) {
            #pragma unroll
            for (int cc = 0; cc < 16; ++cc) {
                int ch = c0 + cc * 16;
                dst[ch * HW_ + m] = qo[ch * HW_ + m];
            }
        }
    }
    __syncthreads();

    // ---- GEMM1: S = (m_in[b] @ q_tile) * (1/16) * mask ----
    // Out-of-range n rows (225..239) are computed on clamped data and simply
    // never read by the softmax -> no guards, uniform EXEC around wmma.
    const float scale = 0.0625f;                       // 1/sqrt(256)
    const int mcol = m0 + col16;
    int xm = 0, ym = 0;
    if (mcol < HW_) { xm = mcol / 15; ym = mcol % 15; }

    for (int rti = 0; rti < 2; ++rti) {
        const int rt = wv + rti * 8;                   // n-tile, uniform per wave
        if (rt >= NT_) break;
        v8f acc = {};
        const int row  = rt * 16 + col16;
        const int rowc = (row < HW_) ? row : (HW_ - 1);    // clamp, stay in-bounds
        const float* arow = m_in + ((size_t)b * HW_ + rowc) * C_;

        #pragma unroll
        for (int ks = 0; ks < 8; ++ks) {
            const int kb = ks * 32 + g * 8;
            BF16Frag af, bf;
            const float4* p4 = (const float4*)(arow + kb);
            float4 a0 = p4[0], a1 = p4[1], a2 = p4[4], a3 = p4[5];
            af.d[0] = pack_bf2(a0.x, a0.y); af.d[1] = pack_bf2(a0.z, a0.w);
            af.d[2] = pack_bf2(a1.x, a1.y); af.d[3] = pack_bf2(a1.z, a1.w);
            af.d[4] = pack_bf2(a2.x, a2.y); af.d[5] = pack_bf2(a2.z, a2.w);
            af.d[6] = pack_bf2(a3.x, a3.y); af.d[7] = pack_bf2(a3.z, a3.w);
            const int off = col16 * 264 + kb;
            bf.q[0] = *(const uint4*)&Bq[off];
            bf.q[1] = *(const uint4*)&Bq[off + 16];
            acc = __builtin_amdgcn_wmma_f32_16x16x32_bf16(
                      false, af.v, false, bf.v, (short)0, acc, false, false);
        }
        // scale + decay mask, stash into LDS for the center-axis softmax
        #pragma unroll
        for (int r = 0; r < 8; ++r) {
            const int n  = rt * 16 + r + 8 * g;
            const int nn = (n < HW_) ? n : 0;
            const int xn = nn / 15, yn = nn % 15;
            const int d  = (xn > xm ? xn - xm : xm - xn) + (yn > ym ? yn - ym : ym - yn);
            S[n * 16 + col16] = acc[r] * scale * dtab[d];
        }
    }
    __syncthreads();

    // ---- softmax over n (225 centers) per column; emit p + bf16 copy ----
    {
        const int col = t & 15, chunk = t >> 4;        // 15 rows per chunk, chunks 0..14
        float lmax = -3.0e38f;
        if (chunk < 15)
            for (int j = 0; j < 15; ++j)
                lmax = fmaxf(lmax, S[(chunk * 15 + j) * 16 + col]);
        red[0][chunk][col] = lmax;
        __syncthreads();
        float cmax = -3.0e38f;
        for (int c = 0; c < 15; ++c) cmax = fmaxf(cmax, red[0][c][col]);
        float lsum = 0.0f;
        if (chunk < 15)
            for (int j = 0; j < 15; ++j) {
                const int n = chunk * 15 + j;
                float e = __expf(S[n * 16 + col] - cmax);
                S[n * 16 + col] = e;
                lsum += e;
            }
        red[1][chunk][col] = lsum;
        __syncthreads();
        float csum = 0.0f;
        for (int c = 0; c < 15; ++c) csum += red[1][c][col];
        const float inv = 1.0f / csum;
        const int m = m0 + col;
        float* pout = out + (size_t)B_ * 512 * HW_ + (size_t)b * HW_ * HW_;
        if (chunk < 15)
            for (int j = 0; j < 15; ++j) {
                const int n = chunk * 15 + j;
                const float pv = S[n * 16 + col] * inv;
                Pbf[col * 264 + n] = f2bf_bits(pv);
                if (m < HW_) pout[(size_t)n * HW_ + m] = pv;
            }
    }

    // ---- GEMM2: mem[v, m] = sum_n m_out[b][n][v] * p[n][m] ----
    // Per k-step: cooperatively stage m_out[n0..n0+31][0..255] into LDS,
    // transposed + bf16-packed; A-frags then read as 2x ds_load_b128.
    {
        const float* mob  = m_out + (size_t)b * HW_ * VD_;
        float*       outm = out + (size_t)b * 512 * HW_;
        v8f acc0 = {}, acc1 = {};
        const int v4 = (t & 63) * 4;                   // staging: 4 v-columns
        const int np = t >> 6;                          // staging: n-pair group 0..3

        for (int ks = 0; ks < 8; ++ks) {
            __syncthreads();                           // prior reads of Slab done
            #pragma unroll
            for (int j = 0; j < 4; ++j) {
                const int pr = np * 4 + j;             // n-pair 0..15 within slab
                const int n0 = ks * 32 + 2 * pr;
                const int n0c = (n0     < HW_) ? n0     : (HW_ - 1);  // clamp: garbage
                const int n1c = (n0 + 1 < HW_) ? n0 + 1 : (HW_ - 1);  // multiplies p==0
                float4 x = *(const float4*)(mob + (size_t)n0c * VD_ + v4);
                float4 y = *(const float4*)(mob + (size_t)n1c * VD_ + v4);
                *(unsigned*)&Slab[(v4 + 0) * 40 + 2 * pr] = pack_bf2(x.x, y.x);
                *(unsigned*)&Slab[(v4 + 1) * 40 + 2 * pr] = pack_bf2(x.y, y.y);
                *(unsigned*)&Slab[(v4 + 2) * 40 + 2 * pr] = pack_bf2(x.z, y.z);
                *(unsigned*)&Slab[(v4 + 3) * 40 + 2 * pr] = pack_bf2(x.w, y.w);
            }
            __syncthreads();

            const int kb = g * 8;
            BF16Frag bfm;
            const int poff = col16 * 264 + ks * 32 + kb;
            bfm.q[0] = *(const uint4*)&Pbf[poff];
            bfm.q[1] = *(const uint4*)&Pbf[poff + 16];

            BF16Frag a0f, a1f;
            const int s0 = (wv * 16 + col16) * 40 + kb;
            const int s1 = ((wv + 8) * 16 + col16) * 40 + kb;
            a0f.q[0] = *(const uint4*)&Slab[s0];
            a0f.q[1] = *(const uint4*)&Slab[s0 + 16];
            a1f.q[0] = *(const uint4*)&Slab[s1];
            a1f.q[1] = *(const uint4*)&Slab[s1 + 16];
            acc0 = __builtin_amdgcn_wmma_f32_16x16x32_bf16(
                       false, a0f.v, false, bfm.v, (short)0, acc0, false, false);
            acc1 = __builtin_amdgcn_wmma_f32_16x16x32_bf16(
                       false, a1f.v, false, bfm.v, (short)0, acc1, false, false);
        }

        const int m = m0 + col16;
        if (m < HW_) {
            #pragma unroll
            for (int r = 0; r < 8; ++r) {
                const int ch0 = wv * 16 + r + 8 * g;
                const int ch1 = (wv + 8) * 16 + r + 8 * g;
                outm[(size_t)ch0 * HW_ + m] = acc0[r];
                outm[(size_t)ch1 * HW_ + m] = acc1[r];
            }
        }
    }
}

extern "C" void kernel_launch(void* const* d_in, const int* in_sizes, int n_in,
                              void* d_out, int out_size, void* d_ws, size_t ws_size,
                              hipStream_t stream) {
    const float* m_in  = (const float*)d_in[0];
    const float* m_out = (const float*)d_in[1];
    const float* q_in  = (const float*)d_in[2];
    const float* q_out = (const float*)d_in[3];
    float* out = (float*)d_out;
    dim3 grid(15, B_);
    spatial_memory_kernel<<<grid, 256, 0, stream>>>(m_in, m_out, q_in, q_out, out);
}
